// GraphRNN_22385369546782
// MI455X (gfx1250) — matrix operand order repaired and checked by reading, not compiled, not run on hardware
//
#include <hip/hip_runtime.h>
#include <math.h>

// Problem sizes (fixed by the reference)
#define BB 64
#define SS 512
#define DD 256
#define HH 128
#define MM 64

typedef _Float16 half4  __attribute__((ext_vector_type(4)));
typedef _Float16 half8  __attribute__((ext_vector_type(8)));
typedef _Float16 half16 __attribute__((ext_vector_type(16)));
typedef float    float4v __attribute__((ext_vector_type(4)));
typedef float    float8 __attribute__((ext_vector_type(8)));

// ---------------------------------------------------------------------------
// Branch-free activations (v_exp_f32 transcendental, no exec-mask branches)
// ---------------------------------------------------------------------------
__device__ __forceinline__ float fast_tanh(float x) {
    x = fminf(10.0f, fmaxf(-10.0f, x));          // tanh(+-10) == +-1 in f32
    const float e = __expf(2.0f * x);
    return __fdividef(e - 1.0f, e + 1.0f);
}

__device__ __forceinline__ float fast_sigmoid(float x) {
    return __fdividef(1.0f, 1.0f + __expf(-x));
}

// ---------------------------------------------------------------------------
// WMMA fragment helpers (V_WMMA_F32_16X16X32_F16 layouts per CDNA5 ISA 7.12.2)
//
// A (16x32 f16, MxK): lanes 0-15 row=lane, halves 0..7 = K 0..7, halves 8..15
// = K 16..23; lanes 16-31 same row, K ranges 8..15 / 24..31. -> two contiguous
// 8-half chunks per lane at column offsets hi*8 and 16+hi*8.
//
// B (32x16 f16, KxN) for out = A @ W^T with W row-major [N_total, K_total]:
// lane n (0-15) holds column n, same K order as A: halves 0..7 = K hi*8..+7,
// halves 8..15 = K 16+hi*8..+7 -> two 8-half chunks of row (colbase+lane16) of W.
// ---------------------------------------------------------------------------
__device__ __forceinline__ half16 load_a_frag(const _Float16* tile, int ld, int lane) {
    const int row = lane & 15;
    const int hi  = lane >> 4;
    const _Float16* p = tile + row * ld + hi * 8;
    union { half16 v; half8 h[2]; } u;
    u.h[0] = *(const half8*)(p);        // K = hi*8 .. hi*8+7
    u.h[1] = *(const half8*)(p + 16);   // K = 16 + hi*8 .. +7
    return u.v;
}

__device__ __forceinline__ half16 load_b_frag(const _Float16* W, int ldk,
                                              int colbase, int kb, int lane) {
    const int col = lane & 15;
    const int hi  = lane >> 4;
    const _Float16* p = W + (size_t)(colbase + col) * ldk + kb + hi * 8;
    union { half16 v; half8 h[2]; } u;
    u.h[0] = *(const half8*)(p);
    u.h[1] = *(const half8*)(p + 16);
    return u.v;
}

__device__ __forceinline__ float8 wmma_acc(half16 a, half16 b, float8 c) {
    // (neg_a, A, neg_b, B, c_mod, C, reuse_a, reuse_b)
    return __builtin_amdgcn_wmma_f32_16x16x32_f16(false, a, false, b,
                                                  (short)0, c, false, false);
}

// ---------------------------------------------------------------------------
// Small utility kernels
// ---------------------------------------------------------------------------
__global__ void f32_to_f16_kernel(const float* __restrict__ src,
                                  _Float16* __restrict__ dst, int n) {
    int i = blockIdx.x * 256 + threadIdx.x;
    if (i < n) dst[i] = (_Float16)src[i];
}

__global__ void zero_f32_kernel(float* __restrict__ p, size_t n) {
    size_t i = (size_t)blockIdx.x * 256 + threadIdx.x;
    if (i < n) p[i] = 0.0f;
}

// ---------------------------------------------------------------------------
// Phase 1: 2-layer graph RNN. 4 blocks x 16 batch rows, 8 waves per block,
// wave w owns hidden columns [16w, 16w+16). Sequential over S=512 steps.
// B-fragments stream from L2-resident f16 weights (hoisted to VGPRs by the
// compiler); x/h tiles live in LDS.
// ---------------------------------------------------------------------------
__global__ __launch_bounds__(256) void graph_rnn_kernel(
    const float* __restrict__ x, const int* __restrict__ mask,
    const _Float16* __restrict__ Wih0h, const _Float16* __restrict__ Whh0h,
    const _Float16* __restrict__ Wih1h, const _Float16* __restrict__ Whh1h,
    const float* __restrict__ gbih0, const float* __restrict__ gbhh0,
    const float* __restrict__ gbih1, const float* __restrict__ gbhh1,
    float* __restrict__ Y)
{
    __shared__ __attribute__((aligned(32))) _Float16 sXt[16 * DD];
    __shared__ __attribute__((aligned(32))) _Float16 sH0[16 * HH];
    __shared__ __attribute__((aligned(32))) _Float16 sH1[16 * HH];
    __shared__ float sB0[HH], sB1[HH];
    __shared__ float sMask[16];

    const int tid  = threadIdx.x;
    const int lane = tid & 31;
    const int col  = lane & 15;
    const int hig  = lane >> 4;
    const int cb   = (tid >> 5) * 16;     // this wave's hidden column base
    const int b0   = blockIdx.x * 16;     // batch-row base

    for (int i = tid; i < HH; i += 256) {
        sB0[i] = gbih0[i] + gbhh0[i];
        sB1[i] = gbih1[i] + gbhh1[i];
    }
    for (int i = tid; i < 16 * HH; i += 256) {
        sH0[i] = (_Float16)0.0f;
        sH1[i] = (_Float16)0.0f;
    }
    __syncthreads();

    for (int t = 0; t < SS; ++t) {
        // stage x[:, t, :] tile (16 x 256) f32 -> f16, vectorized 4-wide
        #pragma unroll
        for (int idx = tid; idx < 16 * DD / 4; idx += 256) {
            const int r = idx >> 6, c4 = idx & 63;
            float4v v = *(const float4v*)(x + ((size_t)(b0 + r) * SS + t) * DD + c4 * 4);
            half4 h;
            h[0] = (_Float16)v[0]; h[1] = (_Float16)v[1];
            h[2] = (_Float16)v[2]; h[3] = (_Float16)v[3];
            *(half4*)(sXt + r * DD + c4 * 4) = h;
        }
        if (tid < 16) sMask[tid] = (float)mask[(b0 + tid) * SS + t];
        __syncthreads();

        // ---- layer 0: tanh(x Wih0^T + h0 Whh0^T + b0)
        float8 acc = {};
        #pragma unroll
        for (int kb = 0; kb < DD; kb += 32)
            acc = wmma_acc(load_a_frag(sXt + kb, DD, lane),
                           load_b_frag(Wih0h, DD, cb, kb, lane), acc);
        #pragma unroll
        for (int kb = 0; kb < HH; kb += 32)
            acc = wmma_acc(load_a_frag(sH0 + kb, HH, lane),
                           load_b_frag(Whh0h, HH, cb, kb, lane), acc);
        __syncthreads();                 // all reads of old h0 done
        {
            const float b = sB0[cb + col];
            #pragma unroll
            for (int r = 0; r < 8; ++r)
                sH0[(hig * 8 + r) * HH + cb + col] = (_Float16)fast_tanh(acc[r] + b);
        }
        __syncthreads();

        // ---- layer 1: tanh(n0 Wih1^T + h1 Whh1^T + b1)
        float8 acc1 = {};
        #pragma unroll
        for (int kb = 0; kb < HH; kb += 32) {
            acc1 = wmma_acc(load_a_frag(sH0 + kb, HH, lane),
                            load_b_frag(Wih1h, HH, cb, kb, lane), acc1);
            acc1 = wmma_acc(load_a_frag(sH1 + kb, HH, lane),
                            load_b_frag(Whh1h, HH, cb, kb, lane), acc1);
        }
        __syncthreads();                 // all reads of old h1 done
        {
            const float b = sB1[cb + col];
            #pragma unroll
            for (int r = 0; r < 8; ++r) {
                const int row = hig * 8 + r;
                const float v = fast_tanh(acc1[r] + b);
                sH1[row * HH + cb + col] = (_Float16)v;   // recurrence: unmasked
                Y[((size_t)(b0 + row) * SS + t) * HH + cb + col] = v * sMask[row];
            }
        }
        __syncthreads();
    }
}

// ---------------------------------------------------------------------------
// Phase 2: autoregressive edge RNN. 2048 blocks x 16 node-rows, 8 waves.
// All edge weights staged once in LDS (~132 KB of 320 KB), 64-step scan.
// Classifier done as a WMMA against a one-hot-row cls matrix; banded
// adjacency scatter fused at the tail of each step.
// ---------------------------------------------------------------------------
__global__ __launch_bounds__(256) void edge_rnn_kernel(
    const float* __restrict__ Y,        // [N, HH] graph state (masked), seeds h0
    const float* __restrict__ eWih0, const float* __restrict__ eWhh0,
    const float* __restrict__ eWih1, const float* __restrict__ eWhh1,
    const float* __restrict__ ebih0, const float* __restrict__ ebhh0,
    const float* __restrict__ ebih1, const float* __restrict__ ebhh1,
    const float* __restrict__ clsW, const float* __restrict__ clsb,
    float* __restrict__ adj)            // [BB, SS, SS], pre-zeroed
{
    __shared__ __attribute__((aligned(32))) _Float16 sWih0[HH * MM];  // 16 KB
    __shared__ __attribute__((aligned(32))) _Float16 sWhh0[HH * HH];  // 32 KB
    __shared__ __attribute__((aligned(32))) _Float16 sWih1[HH * HH];  // 32 KB
    __shared__ __attribute__((aligned(32))) _Float16 sWhh1[HH * HH];  // 32 KB
    __shared__ __attribute__((aligned(32))) _Float16 sClsB[16 * HH];  //  4 KB (row0 = clsW)
    __shared__ __attribute__((aligned(32))) _Float16 sA [16 * MM];    //  2 KB
    __shared__ __attribute__((aligned(32))) _Float16 sH0[16 * HH];    //  4 KB
    __shared__ __attribute__((aligned(32))) _Float16 sH1[16 * HH];    //  4 KB
    __shared__ float sB0[HH], sB1[HH];

    const int tid  = threadIdx.x;
    const int wave = tid >> 5;
    const int lane = tid & 31;
    const int col  = lane & 15;
    const int hig  = lane >> 4;
    const int cb   = wave * 16;
    const int nb   = blockIdx.x * 16;     // global node base

    // stage weights f32 -> f16 into LDS
    for (int i = tid; i < HH * MM; i += 256) sWih0[i] = (_Float16)eWih0[i];
    for (int i = tid; i < HH * HH; i += 256) {
        sWhh0[i] = (_Float16)eWhh0[i];
        sWih1[i] = (_Float16)eWih1[i];
        sWhh1[i] = (_Float16)eWhh1[i];
    }
    for (int i = tid; i < HH; i += 256) {
        sB0[i] = ebih0[i] + ebhh0[i];
        sB1[i] = ebih1[i] + ebhh1[i];
    }
    // classifier B matrix: W row-major [16, HH], row 0 = cls_W, rows 1..15 = 0
    for (int i = tid; i < 16 * HH; i += 256)
        sClsB[i] = (i < HH) ? (_Float16)clsW[i] : (_Float16)0.0f;
    for (int i = tid; i < 16 * MM; i += 256) sA[i] = (_Float16)0.0f;
    for (int i = tid; i < 16 * HH; i += 256) {
        const int r = i >> 7, k = i & (HH - 1);
        sH0[i] = (_Float16)Y[(size_t)(nb + r) * HH + k];  // gs seeds layer-0 h
        sH1[i] = (_Float16)0.0f;
    }
    const float clsb0 = clsb[0];
    __syncthreads();

    for (int j = 0; j < MM; ++j) {
        // ---- layer 0: tanh(A Wih0^T + h0 Whh0^T + b0)
        float8 acc = {};
        #pragma unroll
        for (int kb = 0; kb < MM; kb += 32)
            acc = wmma_acc(load_a_frag(sA + kb, MM, lane),
                           load_b_frag(sWih0, MM, cb, kb, lane), acc);
        #pragma unroll
        for (int kb = 0; kb < HH; kb += 32)
            acc = wmma_acc(load_a_frag(sH0 + kb, HH, lane),
                           load_b_frag(sWhh0, HH, cb, kb, lane), acc);
        __syncthreads();
        {
            const float b = sB0[cb + col];
            #pragma unroll
            for (int r = 0; r < 8; ++r)
                sH0[(hig * 8 + r) * HH + cb + col] = (_Float16)fast_tanh(acc[r] + b);
        }
        __syncthreads();

        // ---- layer 1: tanh(n0 Wih1^T + h1 Whh1^T + b1)
        float8 acc1 = {};
        #pragma unroll
        for (int kb = 0; kb < HH; kb += 32) {
            acc1 = wmma_acc(load_a_frag(sH0 + kb, HH, lane),
                            load_b_frag(sWih1, HH, cb, kb, lane), acc1);
            acc1 = wmma_acc(load_a_frag(sH1 + kb, HH, lane),
                            load_b_frag(sWhh1, HH, cb, kb, lane), acc1);
        }
        __syncthreads();
        {
            const float b = sB1[cb + col];
            #pragma unroll
            for (int r = 0; r < 8; ++r)
                sH1[(hig * 8 + r) * HH + cb + col] = (_Float16)fast_tanh(acc1[r] + b);
        }
        __syncthreads();

        // ---- classifier as WMMA (wave 0): n1[16,HH] @ clsB^T[HH,16], col 0 used
        if (wave == 0) {
            float8 accc = {};
            #pragma unroll
            for (int kb = 0; kb < HH; kb += 32)
                accc = wmma_acc(load_a_frag(sH1 + kb, HH, lane),
                                load_b_frag(sClsB, HH, 0, kb, lane), accc);
            // D layout: lane col==0 (lanes 0 and 16) holds rows hig*8 + r
            if (col == 0) {
                #pragma unroll
                for (int r = 0; r < 8; ++r) {
                    const int row   = hig * 8 + r;
                    const float s   = fast_sigmoid(accc[r] + clsb0);
                    const int node  = nb + row;
                    const int iseq  = node & (SS - 1);     // position in sequence
                    const int mr    = iseq < MM ? iseq : MM;
                    const float val = (j < mr) ? s : 0.0f;
                    sA[row * MM + j] = (_Float16)val;
                    if (j < mr) {
                        const int bidx = node >> 9;        // node / SS
                        const int c    = (iseq > MM ? iseq - MM : 0) + j;
                        // adj (already transposed): adj[b][c][iseq]
                        adj[(size_t)bidx * SS * SS + (size_t)c * SS + iseq] = val;
                    }
                }
            }
        }
        __syncthreads();
    }
}

// ---------------------------------------------------------------------------
// Launch: cvt graph weights -> f16 (ws), zero adj, graph RNN, edge RNN.
// ---------------------------------------------------------------------------
extern "C" void kernel_launch(void* const* d_in, const int* in_sizes, int n_in,
                              void* d_out, int out_size, void* d_ws, size_t ws_size,
                              hipStream_t stream) {
    const float* x     = (const float*)d_in[0];
    const int*   mask  = (const int*)  d_in[1];
    const float* gWih0 = (const float*)d_in[2];
    const float* gWhh0 = (const float*)d_in[3];
    const float* gbih0 = (const float*)d_in[4];
    const float* gbhh0 = (const float*)d_in[5];
    const float* gWih1 = (const float*)d_in[6];
    const float* gWhh1 = (const float*)d_in[7];
    const float* gbih1 = (const float*)d_in[8];
    const float* gbhh1 = (const float*)d_in[9];
    const float* eWih0 = (const float*)d_in[10];
    const float* eWhh0 = (const float*)d_in[11];
    const float* ebih0 = (const float*)d_in[12];
    const float* ebhh0 = (const float*)d_in[13];
    const float* eWih1 = (const float*)d_in[14];
    const float* eWhh1 = (const float*)d_in[15];
    const float* ebih1 = (const float*)d_in[16];
    const float* ebhh1 = (const float*)d_in[17];
    const float* clsW  = (const float*)d_in[18];
    const float* clsb  = (const float*)d_in[19];

    float* Y   = (float*)d_out;                              // [B,S,HG]
    float* adj = (float*)d_out + (size_t)BB * SS * HH;       // [B,S,S]

    // f16 graph weights live in the workspace (~160 KB)
    _Float16* gWih0h = (_Float16*)d_ws;
    _Float16* gWhh0h = gWih0h + HH * DD;
    _Float16* gWih1h = gWhh0h + HH * HH;
    _Float16* gWhh1h = gWih1h + HH * HH;

    f32_to_f16_kernel<<<(HH * DD + 255) / 256, 256, 0, stream>>>(gWih0, gWih0h, HH * DD);
    f32_to_f16_kernel<<<(HH * HH + 255) / 256, 256, 0, stream>>>(gWhh0, gWhh0h, HH * HH);
    f32_to_f16_kernel<<<(HH * HH + 255) / 256, 256, 0, stream>>>(gWih1, gWih1h, HH * HH);
    f32_to_f16_kernel<<<(HH * HH + 255) / 256, 256, 0, stream>>>(gWhh1, gWhh1h, HH * HH);

    const size_t adj_n = (size_t)BB * SS * SS;
    zero_f32_kernel<<<(unsigned)((adj_n + 255) / 256), 256, 0, stream>>>(adj, adj_n);

    graph_rnn_kernel<<<BB / 16, 256, 0, stream>>>(
        x, mask, gWih0h, gWhh0h, gWih1h, gWhh1h,
        gbih0, gbhh0, gbih1, gbhh1, Y);

    edge_rnn_kernel<<<(BB * SS) / 16, 256, 0, stream>>>(
        Y, eWih0, eWhh0, eWih1, eWhh1,
        ebih0, ebhh0, ebih1, ebhh1, clsW, clsb, adj);
}
